// AdaptiveReversibleAttention_55851754717317
// MI455X (gfx1250) — matrix operand order repaired and checked by reading, not compiled, not run on hardware
//
#include <hip/hip_runtime.h>
#include <math.h>

// ---------------------------------------------------------------------------
// AdaptiveReversibleAttention for MI455X (gfx1250), wave32 + WMMA bf16.
// B=1, T=2048, D=1024, H=16, HD=64.  Compute-bound -> all GEMM-shaped work
// runs on v_wmma_f32_16x16x32_bf16 with fp32 accumulation.
// Every tensor is converted fp32->bf16 ONCE (V-operands stored transposed),
// so the GEMM K-loop is pure global_load_b128 + v_wmma: no conversion VALU
// in the hot loop and half the memory traffic of fp32 operands.
// ---------------------------------------------------------------------------

typedef __attribute__((ext_vector_type(16))) __bf16        v16bf;
typedef __attribute__((ext_vector_type(8)))  __bf16        v8bf;
typedef __attribute__((ext_vector_type(8)))  float         v8f;
typedef __attribute__((ext_vector_type(4)))  unsigned int  v4u;

static constexpr int TT  = 2048;  // sequence length
static constexpr int DD  = 1024;  // model dim
static constexpr int NH  = 16;    // heads
static constexpr int HDQ = 64;    // head dim (16-head branches)

typedef unsigned short u16;

__device__ __forceinline__ u16 f32_to_bf16_bits(float f) {
  union { float f; unsigned int u; } in;
  in.f = f;
  unsigned int u = in.u;
  u += 0x7FFFu + ((u >> 16) & 1u);      // round-to-nearest-even
  return (u16)(u >> 16);
}

__device__ __forceinline__ v8bf ld8(const u16* p) {   // 16B-aligned b128 load
  v4u raw = *(const v4u*)p;
  return __builtin_bit_cast(v8bf, raw);
}

// ---------------------------------------------------------------------------
// bf16 GEMM, B transposed:  C[M,N] = epilogue( scale * A[M,K] @ B^T ),
// A: bf16 [M,K] lda, B: bf16 [N,K] ldb.  One wave -> 16(M) x 64(N) strip,
// A fragment reused across 4 B fragments -> 4 WMMA per 32-wide K step.
// outmode: 0 = fp32 C[m*ldc+col] (+bias, *osc, +=) ; 1 = bf16 C[m*ldc+col] ;
//          2 = bf16 transposed C[col*ldc+m]  (both bf16 modes apply bias).
// Requires M%16==0, N%64==0, K%32==0 and 8-element-aligned operand offsets.
// ---------------------------------------------------------------------------
__global__ __launch_bounds__(32)
void gemm_bf16_tn(const u16* __restrict__ A, int lda,
                  const u16* __restrict__ B, int ldb,
                  void* __restrict__ Cv, int ldc,
                  int K, float scale,
                  const float* __restrict__ bias,
                  const float* __restrict__ osc_ptr, int osc_idx,
                  int accumulate, int outmode) {
  const int lane = threadIdx.x;
  const int g = lane >> 4, l = lane & 15;
  const int mBase   = blockIdx.y * 16;
  const int colBase = blockIdx.x * 64;
  const long aOff = (long)(mBase + l) * lda;

  v8f acc[4] = {};
  for (int kt = 0; kt < K; kt += 32) {
    // A frag: elements 0..7 = K 8g..8g+7, 8..15 = K 16+8g..16+8g+7
    const u16* pa = A + aOff + kt + 8 * g;
    v16bf af = __builtin_shufflevector(ld8(pa), ld8(pa + 16),
                                       0, 1, 2, 3, 4, 5, 6, 7,
                                       8, 9, 10, 11, 12, 13, 14, 15);
#pragma unroll
    for (int tn = 0; tn < 4; ++tn) {
      // B frag: elements 0..7 = K 8g..8g+7, 8..15 = K 16+8g..16+8g+7 for col tn*16+l
      const u16* pb = B + (long)(colBase + tn * 16 + l) * ldb + kt + 8 * g;
      v16bf bf = __builtin_shufflevector(ld8(pb), ld8(pb + 16),
                                         0, 1, 2, 3, 4, 5, 6, 7,
                                         8, 9, 10, 11, 12, 13, 14, 15);
      acc[tn] = __builtin_amdgcn_wmma_f32_16x16x32_bf16(
          false, af, false, bf, (short)0, acc[tn], false, false);
    }
  }

  const float osc = osc_ptr ? osc_ptr[osc_idx] : 1.0f;
#pragma unroll
  for (int tn = 0; tn < 4; ++tn) {
    int col = colBase + tn * 16 + l;
    float bv = bias ? bias[col] : 0.0f;
#pragma unroll
    for (int r = 0; r < 8; ++r) {
      int   m   = mBase + g * 8 + r;     // C layout: VGPR r holds M = 8g+r
      float val = osc * (acc[tn][r] * scale + bv);
      if (outmode == 0) {
        float* C = (float*)Cv;
        long idx = (long)m * ldc + col;
        C[idx] = accumulate ? (C[idx] + val) : val;
      } else if (outmode == 1) {
        ((u16*)Cv)[(long)m * ldc + col] = f32_to_bf16_bits(val);
      } else {
        ((u16*)Cv)[(long)col * ldc + m] = f32_to_bf16_bits(val);
      }
    }
  }
}

// ---------------------------------------------------------------------------
// Row softmax: reads fp32 scores S[row,0..ncols), writes bf16 probs Pb and
// optionally accumulates c*prob into fp32 Wout (c = sp_ptr[sp_idx]*invH).
// mode: 0 plain; 1 mask = (param[pidx]>0); 2 gate = sigmoid(param[pidx]).
// ---------------------------------------------------------------------------
__global__ void softmax_rows(float* __restrict__ S, int ncols,
                             int mode, const float* __restrict__ param, int pidx,
                             u16* __restrict__ Pb,
                             float* __restrict__ Wout,
                             const float* __restrict__ sp_ptr, int sp_idx,
                             float invH) {
  __shared__ float red[256];
  const int row = blockIdx.x;
  float* Srow = S + (long)row * ncols;

  float mgate = 1.0f;
  if (mode == 1)      mgate = (param[pidx] > 0.0f) ? 1.0f : 0.0f;
  else if (mode == 2) mgate = 1.0f / (1.0f + expf(-param[pidx]));

  float mx = -3.402823466e38f;
  for (int j = threadIdx.x; j < ncols; j += blockDim.x)
    mx = fmaxf(mx, Srow[j] * mgate);
  red[threadIdx.x] = mx;
  __syncthreads();
  for (int s = blockDim.x / 2; s > 0; s >>= 1) {
    if (threadIdx.x < s) red[threadIdx.x] = fmaxf(red[threadIdx.x], red[threadIdx.x + s]);
    __syncthreads();
  }
  mx = red[0];
  __syncthreads();

  float sum = 0.0f;
  for (int j = threadIdx.x; j < ncols; j += blockDim.x) {
    float e = expf(Srow[j] * mgate - mx);
    Srow[j] = e;
    sum += e;
  }
  red[threadIdx.x] = sum;
  __syncthreads();
  for (int s = blockDim.x / 2; s > 0; s >>= 1) {
    if (threadIdx.x < s) red[threadIdx.x] += red[threadIdx.x + s];
    __syncthreads();
  }
  const float inv = 1.0f / red[0];
  const float c   = (Wout && sp_ptr) ? sp_ptr[sp_idx] * invH : 0.0f;
  for (int j = threadIdx.x; j < ncols; j += blockDim.x) {
    float p = Srow[j] * inv;
    Pb[(long)row * ncols + j] = f32_to_bf16_bits(p);
    if (Wout) Wout[(long)row * ncols + j] += c * p;
  }
}

// elementwise fp32 -> bf16 (n % 4 == 0), vectorized b128 in / b64 out
__global__ void cvt_bf16(const float* __restrict__ in, u16* __restrict__ out, long n) {
  long i = ((long)blockIdx.x * blockDim.x + threadIdx.x) * 4;
  if (i >= n) return;
  float4 v = *(const float4*)(in + i);
  out[i]     = f32_to_bf16_bits(v.x);
  out[i + 1] = f32_to_bf16_bits(v.y);
  out[i + 2] = f32_to_bf16_bits(v.z);
  out[i + 3] = f32_to_bf16_bits(v.w);
}

// batched transpose + convert: out[b][c*rows + r] = in[b][r*cols + c]
__global__ void cvt_bf16_tr(const float* __restrict__ in, u16* __restrict__ out,
                            int rows, int cols) {
  long idx = (long)blockIdx.x * blockDim.x + threadIdx.x;
  long per = (long)rows * cols;
  long b = idx / per, rem = idx % per;
  int  r = (int)(rem / cols), c = (int)(rem % cols);
  out[b * per + (long)c * rows + r] = f32_to_bf16_bits(in[idx]);
}

__global__ void zero_f32(float* __restrict__ p, long n) {
  long i = (long)blockIdx.x * blockDim.x + threadIdx.x;
  if (i < n) p[i] = 0.0f;
}

// F.interpolate(linear, align_corners=False): fp32 [256,D] -> bf16 [2048,D]
__global__ void interp_rows(const float* __restrict__ g, u16* __restrict__ out) {
  int idx = blockIdx.x * blockDim.x + threadIdx.x;
  if (idx >= TT * DD) return;
  int t = idx / DD, c = idx % DD;
  float src = (t + 0.5f) * (256.0f / (float)TT) - 0.5f;
  src = fmaxf(src, 0.0f);
  int i0 = (int)floorf(src); if (i0 > 255) i0 = 255;
  int i1 = i0 + 1;           if (i1 > 255) i1 = 255;
  float w = src - (float)i0;
  float val = g[(long)i0 * DD + c] * (1.0f - w) + g[(long)i1 * DD + c] * w;
  out[idx] = f32_to_bf16_bits(val);
}

extern "C" void kernel_launch(void* const* d_in, const int* in_sizes, int n_in,
                              void* d_out, int out_size, void* d_ws, size_t ws_size,
                              hipStream_t stream) {
  (void)in_sizes; (void)n_in; (void)ws_size;
  const float* q_in   = (const float*)d_in[0];
  const float* k_in   = (const float*)d_in[1];
  const float* v_in   = (const float*)d_in[2];
  const float* sp     = (const float*)d_in[3];   // strategy_probs [4]
  const float* lin_iw = (const float*)d_in[4];
  const float* lin_ib = (const float*)d_in[5];
  const float* lin_ow = (const float*)d_in[6];
  const float* lin_ob = (const float*)d_in[7];
  const float* spat   = (const float*)d_in[8];   // sparsity_pattern [16]
  const float* loc_iw = (const float*)d_in[9];
  const float* loc_ib = (const float*)d_in[10];
  const float* loc_ow = (const float*)d_in[11];
  const float* loc_ob = (const float*)d_in[12];
  const float* glb_iw = (const float*)d_in[13];
  const float* glb_ib = (const float*)d_in[14];
  const float* glb_ow = (const float*)d_in[15];
  const float* glb_ob = (const float*)d_in[16];
  const float* fus_w  = (const float*)d_in[17];  // [D, 2D]
  const float* fus_b  = (const float*)d_in[18];
  const float* qrot   = (const float*)d_in[19];  // [H, 64, 64]
  const float* egate  = (const float*)d_in[20];  // [H, 2, 2]

  float* out0 = (float*)d_out;               // [T, D]
  float* outW = out0 + (long)TT * DD;        // [T, T] combined weights

  // ---------------- workspace layout ----------------
  const long TD = (long)TT * DD, TT2 = (long)TT * TT;
  float* scores = (float*)d_ws;              // [T, T] fp32
  float* gsmall = scores + TT2;              // [256, D] fp32
  u16*   u      = (u16*)(gsmall + 256L * DD);
  u16 *qb, *kb, *vb, *vbT, *Qpb, *Kpb, *VpT, *ctxb, *loutb, *goutb, *qrb, *krb,
      *probs, *wlin_i, *wlin_o, *wloc_i, *wloc_o, *wglb_i, *wglb_o, *wfus, *wrotT;
  qb   = u;           u += TD;     // bf16 query [T,D]
  kb   = u;           u += TD;     // bf16 key   [T,D]
  vb   = u;           u += TD;     // bf16 value [T,D]
  vbT  = u;           u += TD;     // bf16 value^T [D,T]
  Qpb  = u;           u += TD;     // projected Q  [T,D] (or [256,D])
  Kpb  = u;           u += TD;
  VpT  = u;           u += TD;     // projected V^T [D,T] (or [D,256])
  ctxb = u;           u += TD;     // attention context
  loutb= u;           u += TD;     // local branch out
  goutb= u;           u += TD;     // interpolated global out
  qrb  = u;           u += TD;     // rotated q
  krb  = u;           u += TD;     // rotated k
  probs= u;           u += TT2;    // bf16 attention probs [T,T]
  wlin_i = u;         u += 3L*DD*DD;
  wlin_o = u;         u += (long)DD*DD;
  wloc_i = u;         u += 3L*DD*DD;
  wloc_o = u;         u += (long)DD*DD;
  wglb_i = u;         u += 3L*DD*DD;
  wglb_o = u;         u += (long)DD*DD;
  wfus   = u;         u += 2L*DD*DD;
  wrotT  = u;         u += (long)NH*HDQ*HDQ;

  const float scl64  = 0.125f;               // 1/sqrt(64)
  const float scl128 = 0.08838834764831845f; // 1/sqrt(128)
  const float invH   = 1.0f / 16.0f;
  enum { F32 = 0, BF16 = 1, BF16T = 2 };

  auto gemm = [&](const u16* A, int lda, const u16* B, int ldb,
                  void* C, int ldc, int M, int N, int K, float scale,
                  const float* bias, const float* oscp, int osci, int acc,
                  int outmode) {
    dim3 grid(N / 64, M / 16);
    gemm_bf16_tn<<<grid, dim3(32), 0, stream>>>(A, lda, B, ldb, C, ldc, K,
                                                scale, bias, oscp, osci, acc,
                                                outmode);
  };
  auto cvt = [&](const float* in, u16* out, long n) {
    cvt_bf16<<<dim3((unsigned)((n / 4 + 255) / 256)), dim3(256), 0, stream>>>(in, out, n);
  };

  // ---------------- one-time conversions ----------------
  cvt(q_in, qb, TD);  cvt(k_in, kb, TD);  cvt(v_in, vb, TD);
  cvt_bf16_tr<<<dim3((unsigned)((TD + 255) / 256)), dim3(256), 0, stream>>>(v_in, vbT, TT, DD);
  cvt(lin_iw, wlin_i, 3L*DD*DD);  cvt(lin_ow, wlin_o, (long)DD*DD);
  cvt(loc_iw, wloc_i, 3L*DD*DD);  cvt(loc_ow, wloc_o, (long)DD*DD);
  cvt(glb_iw, wglb_i, 3L*DD*DD);  cvt(glb_ow, wglb_o, (long)DD*DD);
  cvt(fus_w,  wfus,   2L*DD*DD);
  cvt_bf16_tr<<<dim3((unsigned)(((long)NH*HDQ*HDQ + 255) / 256)), dim3(256), 0, stream>>>(qrot, wrotT, HDQ, HDQ);
  {
    long n = (long)out_size;
    zero_f32<<<dim3((unsigned)((n + 255) / 256)), dim3(256), 0, stream>>>(out0, n);
  }

  // =========================== linear branch (sp[0]) =======================
  gemm(qb, DD, wlin_i,            DD, Qpb, DD, TT, DD, DD, 1.f, lin_ib,        nullptr, 0, 0, BF16);
  gemm(kb, DD, wlin_i + 1L*DD*DD, DD, Kpb, DD, TT, DD, DD, 1.f, lin_ib + DD,   nullptr, 0, 0, BF16);
  gemm(vb, DD, wlin_i + 2L*DD*DD, DD, VpT, TT, TT, DD, DD, 1.f, lin_ib + 2*DD, nullptr, 0, 0, BF16T);
  for (int h = 0; h < NH; ++h) {
    gemm(Qpb + h*HDQ, DD, Kpb + h*HDQ, DD, scores, TT, TT, TT, HDQ, scl64, nullptr, nullptr, 0, 0, F32);
    softmax_rows<<<dim3(TT), dim3(256), 0, stream>>>(scores, TT, 0, nullptr, 0, probs, outW, sp, 0, invH);
    gemm(probs, TT, VpT + (long)h*HDQ*TT, TT, ctxb + h*HDQ, DD, TT, HDQ, TT, 1.f, nullptr, nullptr, 0, 0, BF16);
  }
  gemm(ctxb, DD, wlin_o, DD, out0, DD, TT, DD, DD, 1.f, lin_ob, sp, 0, 1, F32);

  // =========================== sparse branch (sp[1]) =======================
  for (int h = 0; h < NH; ++h) {
    gemm(qb + h*HDQ, DD, kb + h*HDQ, DD, scores, TT, TT, TT, HDQ, scl64, nullptr, nullptr, 0, 0, F32);
    softmax_rows<<<dim3(TT), dim3(256), 0, stream>>>(scores, TT, 1, spat, h, probs, outW, sp, 1, invH);
    gemm(probs, TT, vbT + (long)h*HDQ*TT, TT, out0 + h*HDQ, DD, TT, HDQ, TT, 1.f, nullptr, sp, 1, 1, F32);
  }

  // ======================= hierarchical: local (8 heads, hd=128) ===========
  gemm(qb, DD, wloc_i,            DD, Qpb, DD, TT, DD, DD, 1.f, loc_ib,        nullptr, 0, 0, BF16);
  gemm(kb, DD, wloc_i + 1L*DD*DD, DD, Kpb, DD, TT, DD, DD, 1.f, loc_ib + DD,   nullptr, 0, 0, BF16);
  gemm(vb, DD, wloc_i + 2L*DD*DD, DD, VpT, TT, TT, DD, DD, 1.f, loc_ib + 2*DD, nullptr, 0, 0, BF16T);
  for (int h = 0; h < 8; ++h) {
    gemm(Qpb + h*128, DD, Kpb + h*128, DD, scores, TT, TT, TT, 128, scl128, nullptr, nullptr, 0, 0, F32);
    softmax_rows<<<dim3(TT), dim3(256), 0, stream>>>(scores, TT, 0, nullptr, 0, probs, nullptr, nullptr, 0, 0.f);
    gemm(probs, TT, VpT + (long)h*128*TT, TT, ctxb + h*128, DD, TT, 128, TT, 1.f, nullptr, nullptr, 0, 0, BF16);
  }
  gemm(ctxb, DD, wloc_o, DD, loutb, DD, TT, DD, DD, 1.f, loc_ob, nullptr, 0, 0, BF16);

  // ============== hierarchical: global (stride-8 rows, T=256) ==============
  gemm(qb, 8*DD, wglb_i,            DD, Qpb, DD, 256, DD, DD, 1.f, glb_ib,        nullptr, 0, 0, BF16);
  gemm(kb, 8*DD, wglb_i + 1L*DD*DD, DD, Kpb, DD, 256, DD, DD, 1.f, glb_ib + DD,   nullptr, 0, 0, BF16);
  gemm(vb, 8*DD, wglb_i + 2L*DD*DD, DD, VpT, 256, 256, DD, DD, 1.f, glb_ib + 2*DD, nullptr, 0, 0, BF16T);
  for (int h = 0; h < 8; ++h) {
    gemm(Qpb + h*128, DD, Kpb + h*128, DD, scores, 256, 256, 256, 128, scl128, nullptr, nullptr, 0, 0, F32);
    softmax_rows<<<dim3(256), dim3(256), 0, stream>>>(scores, 256, 0, nullptr, 0, probs, nullptr, nullptr, 0, 0.f);
    gemm(probs, 256, VpT + (long)h*128*256, 256, ctxb + h*128, DD, 256, 128, 256, 1.f, nullptr, nullptr, 0, 0, BF16);
  }
  gemm(ctxb, DD, wglb_o, DD, gsmall, DD, 256, DD, DD, 1.f, glb_ob, nullptr, 0, 0, F32);
  interp_rows<<<dim3((TT*DD + 255) / 256), dim3(256), 0, stream>>>(gsmall, goutb);

  // fusion: out += sp[2] * (cat(local, global) @ fus_w^T + fus_b)
  gemm(loutb, DD, wfus,      2*DD, out0, DD, TT, DD, DD, 1.f, fus_b,   sp, 2, 1, F32);
  gemm(goutb, DD, wfus + DD, 2*DD, out0, DD, TT, DD, DD, 1.f, nullptr, sp, 2, 1, F32);

  // =========================== quantum branch (sp[3]) ======================
  for (int h = 0; h < NH; ++h)
    gemm(qb + h*HDQ, DD, wrotT + (long)h*HDQ*HDQ, HDQ, qrb + h*HDQ, DD, TT, HDQ, HDQ, 1.f, nullptr, nullptr, 0, 0, BF16);
  for (int h = 0; h < NH; ++h)
    gemm(kb + h*HDQ, DD, wrotT + (long)h*HDQ*HDQ, HDQ, krb + h*HDQ, DD, TT, HDQ, HDQ, 1.f, nullptr, nullptr, 0, 0, BF16);
  for (int h = 0; h < NH; ++h) {
    gemm(qrb + h*HDQ, DD, krb + h*HDQ, DD, scores, TT, TT, TT, HDQ, scl64, nullptr, nullptr, 0, 0, F32);
    softmax_rows<<<dim3(TT), dim3(256), 0, stream>>>(scores, TT, 2, egate, h*4, probs, outW, sp, 3, invH);
    gemm(probs, TT, vbT + (long)h*HDQ*TT, TT, out0 + h*HDQ, DD, TT, HDQ, TT, 1.f, nullptr, sp, 3, 1, F32);
  }
}
